// PseudoGroupContrast_72292889526452
// MI455X (gfx1250) — compile-verified, matrix operand order checked by reading, not run on hardware
//
#include <hip/hip_runtime.h>

typedef __attribute__((ext_vector_type(16))) __bf16 v16bf;
typedef __attribute__((ext_vector_type(8)))  __bf16 v8bf;
typedef __attribute__((ext_vector_type(4)))  __bf16 v4bf;
typedef __attribute__((ext_vector_type(8)))  float  v8f;
typedef __attribute__((ext_vector_type(4)))  float  v4f;

#define PROJ      128
#define CLS       3
#define QSZ       125
#define NQ        (CLS*QSZ)        // 375
#define NQ_PAD    384
#define NTILES    (NQ_PAD/16)      // 24
#define KTILES    (PROJ/32)        // 4
#define TEMP_INV  2.0f             // 1/0.5
#define EPS_      1e-6f

#define WAVES      4
#define THREADS    (WAVES*32)
#define ROWS_ITER  (WAVES*16)      // 64
#define ITERS      4
#define ROWS_BLOCK (ROWS_ITER*ITERS) // 256

// Build a v16bf A/B fragment from two 16-byte contiguous LDS chunks
// (16-bit WMMA operand layout: per-lane chunk at kbase and kbase+16).
__device__ __forceinline__ v16bf load_frag(const __bf16* p) {
    v8bf lo = *(const v8bf*)p;
    v8bf hi = *(const v8bf*)(p + 16);
    v16bf t;
#pragma unroll
    for (int i = 0; i < 8; ++i) { t[i] = lo[i]; t[i + 8] = hi[i]; }
    return t;
}

__launch_bounds__(THREADS)
__global__ void pgc_main_kernel(const float* __restrict__ act,
                                const float* __restrict__ ema,
                                const float* __restrict__ plab,
                                const float* __restrict__ mask,
                                const float* __restrict__ qlist,
                                float* __restrict__ partials,
                                int Btot)
{
    __shared__ __align__(16) __bf16 q_lds[NQ_PAD * PROJ];       // 96 KB
    __shared__ __align__(16) __bf16 f_lds[WAVES][16 * PROJ];    // 16 KB
    __shared__ float lpos_lds[WAVES][16];
    __shared__ float maskr_lds[WAVES][16];
    __shared__ int   label_lds[WAVES][16];
    __shared__ float wave_part[WAVES];

    const int tid   = threadIdx.x;
    const int wv    = tid >> 5;
    const int lane  = tid & 31;
    const int col   = lane & 15;   // row-of-A-tile / column-of-B-tile
    const int half  = lane >> 4;
    const int kbase = half * 8;

    // ---- stage queue into LDS as bf16 (rows 375..383 zero-padded) ----
    {
        const v4f* q4 = (const v4f*)qlist;
        v4bf* qd = (v4bf*)q_lds;
        for (int i = tid; i < (NQ * PROJ) / 4; i += THREADS) {
            v4f x = q4[i];
            v4bf b;
            b[0] = (__bf16)x[0]; b[1] = (__bf16)x[1];
            b[2] = (__bf16)x[2]; b[3] = (__bf16)x[3];
            qd[i] = b;
        }
        v4bf z = {};
        for (int i = tid; i < ((NQ_PAD - NQ) * PROJ) / 4; i += THREADS)
            qd[(NQ * PROJ) / 4 + i] = z;
    }
    __syncthreads();

    float block_acc = 0.0f;

    for (int it = 0; it < ITERS; ++it) {
        const long rbase = (long)blockIdx.x * ROWS_BLOCK + (long)it * ROWS_ITER + wv * 16;

        // ---- normalize 16 rows; positive pair, label, mask (all f32) ----
        for (int r = 0; r < 16; ++r) {
            long row = rbase + r;
            float mval;
            if (row >= Btot) { row = Btot - 1; mval = 0.0f; }
            else             { mval = mask[row]; }
            const v4f a = *(const v4f*)(act + row * PROJ + lane * 4);
            const v4f e = *(const v4f*)(ema + row * PROJ + lane * 4);
            float aa = a[0]*a[0] + a[1]*a[1] + a[2]*a[2] + a[3]*a[3];
            float ee = e[0]*e[0] + e[1]*e[1] + e[2]*e[2] + e[3]*e[3];
            float ae = a[0]*e[0] + a[1]*e[1] + a[2]*e[2] + a[3]*e[3];
#pragma unroll
            for (int m = 16; m >= 1; m >>= 1) {
                aa += __shfl_xor(aa, m);
                ee += __shfl_xor(ee, m);
                ae += __shfl_xor(ae, m);
            }
            const float sa = 1.0f / fmaxf(sqrtf(aa), 1e-12f);
            const float se = 1.0f / fmaxf(sqrtf(ee), 1e-12f);
            v4bf nb;
            nb[0] = (__bf16)(a[0] * sa); nb[1] = (__bf16)(a[1] * sa);
            nb[2] = (__bf16)(a[2] * sa); nb[3] = (__bf16)(a[3] * sa);
            *(v4bf*)&f_lds[wv][r * PROJ + lane * 4] = nb;
            if (lane == 0) {
                lpos_lds[wv][r] = __expf(ae * sa * se * TEMP_INV);
                const float* p = plab + row * CLS;
                float l0 = p[0], l1 = p[1], l2 = p[2];
                int lb = 0; float bst = l0;
                if (l1 > bst) { bst = l1; lb = 1; }
                if (l2 > bst) { lb = 2; }
                label_lds[wv][r] = lb;
                maskr_lds[wv][r] = mval;
            }
        }

        // ---- A fragments for this wave's 16xK=128 tile (kept in VGPRs) ----
        v16bf Af[KTILES];
#pragma unroll
        for (int kt = 0; kt < KTILES; ++kt)
            Af[kt] = load_frag(&f_lds[wv][col * PROJ + kt * 32 + kbase]);

        // class-block start column for each of this lane's 8 rows
        int q0v[8];
#pragma unroll
        for (int v = 0; v < 8; ++v)
            q0v[v] = label_lds[wv][half * 8 + v] * QSZ;

        // ---- single GEMM pass: accumulate per-row statistics ----
        //   acc  = sum_all  exp(t)          (t = s/T)
        //   tacc = sum_blk  t
        //   wacc = sum_blk  exp(-t)
        //   w2acc= sum_blk  exp(-2t)
        float acc[8], tacc[8], wacc[8], w2acc[8];
#pragma unroll
        for (int v = 0; v < 8; ++v) { acc[v] = 0.f; tacc[v] = 0.f; wacc[v] = 0.f; w2acc[v] = 0.f; }

        for (int nt = 0; nt < NTILES; ++nt) {
            const __bf16* brow = &q_lds[(nt * 16 + col) * PROJ + kbase];
            // Two independent accumulator chains: consecutive WMMAs have no
            // D->C RAW, so the scheduler fills hazard slots with real work
            // instead of v_nop padding (bf16 WMMA->WMMA needs 5 slots).
            v8f C0 = {}, C1 = {};
#pragma unroll
            for (int kt = 0; kt < KTILES; kt += 2) {
                v16bf B0 = load_frag(brow + kt * 32);
                v16bf B1 = load_frag(brow + (kt + 1) * 32);
                C0 = __builtin_amdgcn_wmma_f32_16x16x32_bf16(
                         false, Af[kt],     false, B0, (short)0, C0, false, false);
                C1 = __builtin_amdgcn_wmma_f32_16x16x32_bf16(
                         false, Af[kt + 1], false, B1, (short)0, C1, false, false);
            }
            const int q = nt * 16 + col;
            const float vmask = (q < NQ) ? 1.0f : 0.0f;
#pragma unroll
            for (int v = 0; v < 8; ++v) {
                const float t  = (C0[v] + C1[v]) * TEMP_INV;
                const float ep = __expf(t);
                acc[v] += vmask * ep;
                // in-block? (pad columns auto-excluded: q - q0 >= 125)
                const float inb = ((unsigned)(q - q0v[v]) < (unsigned)QSZ) ? 1.0f : 0.0f;
                const float w  = __builtin_amdgcn_rcpf(ep);   // exp(-t)
                const float wi = inb * w;
                tacc[v]  += inb * t;
                wacc[v]  += wi;
                w2acc[v] += wi * w;
            }
        }
        // xor-reduce the 4 statistic sets over each 16-lane half
#pragma unroll
        for (int v = 0; v < 8; ++v) {
#pragma unroll
            for (int m = 1; m <= 8; m <<= 1) {
                acc[v]   += __shfl_xor(acc[v], m);
                tacc[v]  += __shfl_xor(tacc[v], m);
                wacc[v]  += __shfl_xor(wacc[v], m);
                w2acc[v] += __shfl_xor(w2acc[v], m);
            }
        }

        // ---- per-row epilogue (values replicated across each half) ----
        //   per_pos = QSZ*log(den) - tacc - (c*wacc - 0.5*c^2*w2acc), c = eps*den
        //   per_ema = log(den) - log(lpos + c)
        float s8 = 0.0f;
#pragma unroll
        for (int v = 0; v < 8; ++v) {
            const int m = half * 8 + v;
            const float lp  = lpos_lds[wv][m];
            const float den = acc[v] + lp;
            const float logden = __logf(den);
            const float c   = EPS_ * den;
            const float sum = (float)(QSZ + 1) * logden - tacc[v]
                              - (c * wacc[v] - 0.5f * c * c * w2acc[v])
                              - __logf(lp + c);
            s8 += maskr_lds[wv][m] * sum * (1.0f / (QSZ + 1));
        }
        s8 += __shfl_xor(s8, 16);
        block_acc += s8;
    }

    if (lane == 0) wave_part[wv] = block_acc;
    __syncthreads();
    if (tid == 0) {
        float t = 0.0f;
#pragma unroll
        for (int w = 0; w < WAVES; ++w) t += wave_part[w];
        partials[blockIdx.x] = t;
    }
}

__launch_bounds__(256)
__global__ void pgc_reduce_kernel(const float* __restrict__ partials,
                                  float* __restrict__ out, int n, float invB)
{
    __shared__ float sm[256];
    float s = 0.0f;
    for (int i = threadIdx.x; i < n; i += 256) s += partials[i];
    sm[threadIdx.x] = s;
    __syncthreads();
    for (int off = 128; off > 0; off >>= 1) {
        if (threadIdx.x < off) sm[threadIdx.x] += sm[threadIdx.x + off];
        __syncthreads();
    }
    if (threadIdx.x == 0) out[0] = sm[0] * invB;
}

extern "C" void kernel_launch(void* const* d_in, const int* in_sizes, int n_in,
                              void* d_out, int out_size, void* d_ws, size_t ws_size,
                              hipStream_t stream) {
    const float* act   = (const float*)d_in[0];
    const float* ema   = (const float*)d_in[1];
    const float* plab  = (const float*)d_in[2];
    const float* mask  = (const float*)d_in[3];
    const float* qlist = (const float*)d_in[4];
    const int Btot = in_sizes[3];                 // mask has B elements
    const int nblocks = (Btot + ROWS_BLOCK - 1) / ROWS_BLOCK;
    float* partials = (float*)d_ws;

    pgc_main_kernel<<<nblocks, THREADS, 0, stream>>>(
        act, ema, plab, mask, qlist, partials, Btot);
    pgc_reduce_kernel<<<1, 256, 0, stream>>>(
        partials, (float*)d_out, nblocks, 1.0f / (float)Btot);
}
